// PCAE_21852793602118
// MI455X (gfx1250) — matrix operand order repaired and checked by reading, compiled, not run on hardware
//
#include <hip/hip_runtime.h>
#include <math.h>

typedef __attribute__((ext_vector_type(16))) _Float16 v16h;
typedef __attribute__((ext_vector_type(8)))  float    v8f;
typedef __attribute__((ext_vector_type(8)))  unsigned int v8u;

#define KCAP 24
#define BN   64
#define TS   11
#define HH   40
#define WW   40

// ---------------------------------------------------------------- cast f32->f16
__global__ __launch_bounds__(256) void pcae_cast_f16(const float* __restrict__ in,
                                                     _Float16* __restrict__ out, int n) {
    int i = blockIdx.x * 256 + threadIdx.x;
    if (i < n) out[i] = (_Float16)in[i];
}

// ---------------------------------------------------------------- conv1 (1->128, 3x3 s2) direct
__global__ __launch_bounds__(256) void pcae_conv1(const float* __restrict__ x,
                                                  const float* __restrict__ w1,
                                                  const float* __restrict__ b1,
                                                  _Float16* __restrict__ act1) {
    const int kb = blockIdx.x;            // K*B
    const int k = kb / BN, b = kb % BN;
    const float* xb = x + (size_t)b * (HH * WW);
    for (int idx = threadIdx.x; idx < 128 * 19 * 19; idx += 256) {
        int co = idx / 361;
        int p  = idx - co * 361;
        int oy = p / 19, ox = p - oy * 19;
        const float* wk = w1 + (size_t)(k * 128 + co) * 9;
        float s = b1[k * 128 + co];
        #pragma unroll
        for (int ky = 0; ky < 3; ++ky)
            #pragma unroll
            for (int kx = 0; kx < 3; ++kx)
                s += xb[(2 * oy + ky) * WW + (2 * ox + kx)] * wk[ky * 3 + kx];
        s = s > 0.f ? s : 0.f;
        act1[((size_t)kb * 128 + co) * 361 + p] = (_Float16)s;
    }
}

// ---------------------------------------------------------------- 3x3 conv implicit GEMM, WMMA f16
// M = B*Ho*Wo (multiple of 64), N = 128, Kd = 128*9 = 1152
// CL=true stores channel-last activations [B, Ho*Wo, Cout] (for the 1x1 consumer)
template <int S, int HI, int WI, int HO, int WO, bool CL>
__global__ __launch_bounds__(256) void pcae_conv3x3_wmma(const _Float16* __restrict__ act_in,
                                                         const _Float16* __restrict__ wgt,
                                                         const float* __restrict__ bias,
                                                         _Float16* __restrict__ act_out) {
    constexpr int Cin = 128, Cout = 128, Kd = Cin * 9;
    __shared__ _Float16 As[64 * 32];     // [m][kd]   4 KB
    __shared__ _Float16 BsT[128 * 32];   // [n][kd]   8 KB
    const int k      = blockIdx.y;
    const int mtile  = blockIdx.x * 64;
    const int tid    = threadIdx.x;
    const int lane   = tid & 31;
    const int wave   = tid >> 5;
    const int laneLo = lane & 15, laneHi = lane >> 4;
    const int wm = (wave >> 2) * 32;     // wave M offset in tile
    const int wn = (wave & 3) * 32;      // wave N offset in tile
    const _Float16* ain = act_in + (size_t)k * BN * Cin * HI * WI;
    const _Float16* wk  = wgt + (size_t)k * Cout * Kd;
    v8f acc[2][2] = {};
    for (int kk = 0; kk < Kd; kk += 32) {
        // ---- stage B: async LDS-DMA of the contiguous weight slice (no VGPR round trip)
        {
            int n  = tid >> 1;             // 0..127
            int c0 = (tid & 1) * 16;       // halves
            const _Float16* src = wk + (size_t)n * Kd + kk + c0;
            _Float16* dst = &BsT[n * 32 + c0];
#if defined(__HIP_DEVICE_COMPILE__)
            unsigned lds_off = (unsigned)(unsigned long long)dst;    // flat->LDS: low bits
            unsigned long long gaddr = (unsigned long long)src;
            asm volatile("global_load_async_to_lds_b128 %0, %1, off"
                         :: "v"(lds_off), "v"(gaddr) : "memory");
            asm volatile("global_load_async_to_lds_b128 %0, %1, off"
                         :: "v"(lds_off + 16u), "v"(gaddr + 16ull) : "memory");
#else
            for (int i = 0; i < 16; i += 2)
                *(unsigned int*)&dst[i] = *(const unsigned int*)&src[i];
#endif
            if (kk + 32 < Kd) __builtin_prefetch(src + 32, 0, 1);    // global_prefetch next K-slice
        }
        // ---- stage A (im2col gather): 64 rows x 32 kd
        {
            int r  = tid >> 2;
            int c0 = (tid & 3) * 8;
            int m  = mtile + r;
            int b  = m / (HO * WO);
            int pq = m - b * (HO * WO);
            int oy = pq / WO, ox = pq - oy * WO;
            const _Float16* base = ain + (size_t)b * Cin * (HI * WI);
            #pragma unroll
            for (int i = 0; i < 8; ++i) {
                int kd  = kk + c0 + i;
                int ci  = kd / 9;
                int rem = kd - ci * 9;
                int ky  = rem / 3, kx = rem - ky * 3;
                As[r * 32 + c0 + i] = base[(size_t)ci * (HI * WI) + (oy * S + ky) * WI + (ox * S + kx)];
            }
        }
#if defined(__HIP_DEVICE_COMPILE__)
        asm volatile("s_wait_asynccnt 0x0" ::: "memory");
#endif
        __syncthreads();
        // ---- build fragments (contiguous k = 16*laneHi + s in both A and B)
        v16h afr[2], bfr[2];
        #pragma unroll
        for (int i = 0; i < 2; ++i) {
            v8u au;
            int mrow = wm + i * 16 + laneLo;
            #pragma unroll
            for (int v = 0; v < 8; ++v) {
                int kb2 = laneHi * 16 + 2 * v;
                au[v] = *(const unsigned int*)&As[mrow * 32 + kb2];
            }
            afr[i] = __builtin_bit_cast(v16h, au);
        }
        #pragma unroll
        for (int j = 0; j < 2; ++j) {
            v8u bu;
            int ncol = wn + j * 16 + laneLo;
            int kb2  = laneHi * 16;
            #pragma unroll
            for (int v = 0; v < 8; ++v)
                bu[v] = *(const unsigned int*)&BsT[ncol * 32 + kb2 + 2 * v];
            bfr[j] = __builtin_bit_cast(v16h, bu);
        }
        #pragma unroll
        for (int i = 0; i < 2; ++i)
            #pragma unroll
            for (int j = 0; j < 2; ++j)
                acc[i][j] = __builtin_amdgcn_wmma_f32_16x16x32_f16(
                    false, afr[i], false, bfr[j], (short)0, acc[i][j], false, false);
        __syncthreads();
    }
    // ---- epilogue: bias + relu -> f16 activations
    _Float16* aout = act_out + (size_t)k * BN * Cout * HO * WO;
    #pragma unroll
    for (int i = 0; i < 2; ++i) {
        #pragma unroll
        for (int j = 0; j < 2; ++j) {
            int n    = wn + j * 16 + laneLo;
            float bv = bias[k * Cout + n];
            #pragma unroll
            for (int r = 0; r < 8; ++r) {
                int m  = mtile + wm + i * 16 + (laneHi ? 8 : 0) + r;
                int b  = m / (HO * WO);
                int pq = m - b * (HO * WO);
                float v = acc[i][j][r] + bv;
                v = v > 0.f ? v : 0.f;
                if (CL)
                    aout[((size_t)b * (HO * WO) + pq) * Cout + n] = (_Float16)v;
                else
                    aout[((size_t)b * Cout + n) * (HO * WO) + pq] = (_Float16)v;
            }
        }
    }
}

// ---------------------------------------------------------------- conv5 (1x1, 128->16), WMMA, no relu
// act4 is channel-last [K, B, 25, 128] -> A fragment rows are K-contiguous
__global__ __launch_bounds__(256) void pcae_conv1x1_wmma(const _Float16* __restrict__ act4,
                                                         const _Float16* __restrict__ w5h,
                                                         const float* __restrict__ b5,
                                                         float* __restrict__ out5) {
    constexpr int Cin = 128, Cout = 16, P = 25, Mtot = BN * P;
    const int k      = blockIdx.y;
    const int lane   = threadIdx.x & 31;
    const int wave   = threadIdx.x >> 5;
    const int laneLo = lane & 15, laneHi = lane >> 4;
    const int mbase  = (blockIdx.x * 8 + wave) * 16;
    if (mbase >= Mtot) return;
    const _Float16* ain = act4 + (size_t)k * Mtot * Cin;
    const _Float16* wk  = w5h + (size_t)k * Cout * Cin;
    const int m = mbase + laneLo;                       // row index == b*25+p
    const _Float16* abase = ain + (size_t)m * Cin;
    const _Float16* bbase = wk + (size_t)laneLo * Cin + laneHi * 16;
    v8f acc = {};
    #pragma unroll
    for (int kk = 0; kk < Cin; kk += 32) {
        v8u au, bu;
        #pragma unroll
        for (int v = 0; v < 8; ++v) {
            int kd0 = kk + laneHi * 16 + 2 * v;
            au[v] = *(const unsigned int*)&abase[kd0];
        }
        #pragma unroll
        for (int v = 0; v < 8; ++v)
            bu[v] = *(const unsigned int*)&bbase[kk + 2 * v];
        acc = __builtin_amdgcn_wmma_f32_16x16x32_f16(
            false, __builtin_bit_cast(v16h, au), false, __builtin_bit_cast(v16h, bu),
            (short)0, acc, false, false);
    }
    int n = laneLo;
    float bv = b5[k * Cout + n];
    #pragma unroll
    for (int r = 0; r < 8; ++r) {
        int mo = mbase + (laneHi ? 8 : 0) + r;
        int b = mo / P, p = mo - b * P;
        out5[((size_t)(k * BN + b) * Cout + n) * P + p] = acc[r] + bv;
    }
}

// ---------------------------------------------------------------- attention pool + activations + output packing
__global__ __launch_bounds__(256) void pcae_attention(const float* __restrict__ out5,
                                                      const float* __restrict__ tpl,
                                                      float* __restrict__ d_out,
                                                      float* __restrict__ pose_ws,
                                                      float* __restrict__ dm_ws) {
    int idx = blockIdx.x * 256 + threadIdx.x;
    if (idx >= KCAP * BN) return;
    int k = idx / BN, b = idx - k * BN;
    const float* o = out5 + (size_t)(k * BN + b) * 16 * 25;
    float pooled[15];
    #pragma unroll
    for (int f = 0; f < 15; ++f) pooled[f] = 0.f;
    for (int y = 0; y < 5; ++y) {
        const float* arow = o + 15 * 25 + y * 5;
        float mx = arow[0];
        for (int x2 = 1; x2 < 5; ++x2) mx = fmaxf(mx, arow[x2]);
        float e[5], s = 0.f;
        for (int x2 = 0; x2 < 5; ++x2) { e[x2] = __expf(arow[x2] - mx); s += e[x2]; }
        float inv = 1.f / s;
        for (int x2 = 0; x2 < 5; ++x2) {
            float w = e[x2] * inv;
            #pragma unroll
            for (int f = 0; f < 15; ++f) pooled[f] += o[f * 25 + y * 5 + x2] * w;
        }
    }
    float xm[6], cz[8];
    #pragma unroll
    for (int j = 0; j < 6; ++j) xm[j] = fmaxf(pooled[j], 0.f);
    float dm = 1.f / (1.f + __expf(-pooled[6]));
    #pragma unroll
    for (int j = 0; j < 8; ++j) cz[j] = fmaxf(pooled[7 + j], 0.f);
    const size_t bk = (size_t)b * KCAP + k;
    size_t oa = 1 + bk * 136;                               // input_ocae
    d_out[oa] = dm;
    for (int j = 0; j < 6; ++j) d_out[oa + 1 + j] = xm[j];
    for (int j = 0; j < 121; ++j) d_out[oa + 7 + j] = tpl[k * 121 + j];
    for (int j = 0; j < 8; ++j) d_out[oa + 128 + j] = cz[j];
    size_t ox = 1 + (size_t)BN * KCAP * 136 + bk * 6;       // x_m
    for (int j = 0; j < 6; ++j) d_out[ox + j] = xm[j];
    size_t od = 1 + (size_t)BN * KCAP * 136 + (size_t)BN * KCAP * 6 + bk;  // d_m
    d_out[od] = dm;
    for (int j = 0; j < 6; ++j) pose_ws[bk * 6 + j] = xm[j];
    dm_ws[bk] = dm;
}

// ---------------------------------------------------------------- affine warp of templates (inverse affine + bilinear)
__global__ __launch_bounds__(256) void pcae_warp(const float* __restrict__ pose,
                                                 const float* __restrict__ tpl,
                                                 float* __restrict__ tt) {
    const int bk = blockIdx.x;            // b*K + k
    const int k  = bk % KCAP;
    const float* p = pose + (size_t)bk * 6;
    const float DEG = 3.14159265358979f / 180.f;
    float ang = p[0] * DEG, tx = p[1], ty = p[2];
    float sc  = fmaxf(p[3], 1e-2f);
    float shx = p[4] * DEG, shy = p[5] * DEG;
    float ca = __cosf(ang - shy), sa = __sinf(ang - shy);
    float icsy = 1.f / __cosf(shy), tshx = tanf(shx);
    float a  = ca * icsy;
    float bb = -ca * tshx * icsy - __sinf(ang);
    float c  = sa * icsy;
    float d  = -sa * tshx * icsy + __cosf(ang);
    float isc = 1.f / sc;
    float m0 = d * isc, m1 = -bb * isc, m3 = -c * isc, m4 = a * isc;
    const float cx = (TS - 1) * 0.5f, cy = (TS - 1) * 0.5f;
    float m2 = m0 * (-cx - tx) + m1 * (-cy - ty) + cx;
    float m5 = m3 * (-cx - tx) + m4 * (-cy - ty) + cy;
    const float* t = tpl + (size_t)k * (TS * TS);
    for (int pix = threadIdx.x; pix < HH * WW; pix += 256) {
        int yo = pix / WW, xo = pix - yo * WW;
        float gx = (xo + 0.5f) * ((float)TS / WW) - 0.5f;
        float gy = (yo + 0.5f) * ((float)TS / HH) - 0.5f;
        float xin = m0 * gx + m1 * gy + m2;
        float yin = m3 * gx + m4 * gy + m5;
        float x0 = floorf(xin), y0 = floorf(yin);
        float wx = xin - x0, wy = yin - y0;
        int x0i = (int)x0, y0i = (int)y0;
        float v00 = 0.f, v01 = 0.f, v10 = 0.f, v11 = 0.f;
        bool xa = (x0i >= 0) & (x0i < TS), xb2 = (x0i + 1 >= 0) & (x0i + 1 < TS);
        bool ya = (y0i >= 0) & (y0i < TS), yb2 = (y0i + 1 >= 0) & (y0i + 1 < TS);
        if (ya && xa)  v00 = t[y0i * TS + x0i];
        if (ya && xb2) v01 = t[y0i * TS + x0i + 1];
        if (yb2 && xa) v10 = t[(y0i + 1) * TS + x0i];
        if (yb2 && xb2) v11 = t[(y0i + 1) * TS + x0i + 1];
        tt[(size_t)bk * (HH * WW) + pix] =
            v00 * (1 - wx) * (1 - wy) + v01 * wx * (1 - wy) + v10 * (1 - wx) * wy + v11 * wx * wy;
    }
}

// ---------------------------------------------------------------- per-sample std (ddof=1)
__global__ __launch_bounds__(256) void pcae_std(const float* __restrict__ x, float* __restrict__ std_ws) {
    int b = blockIdx.x;
    __shared__ float ssum[256], ssq[256];
    float s = 0.f, q = 0.f;
    for (int i = threadIdx.x; i < HH * WW; i += 256) {
        float v = x[(size_t)b * (HH * WW) + i];
        s += v; q += v * v;
    }
    ssum[threadIdx.x] = s; ssq[threadIdx.x] = q;
    __syncthreads();
    for (int st = 128; st > 0; st >>= 1) {
        if (threadIdx.x < st) { ssum[threadIdx.x] += ssum[threadIdx.x + st]; ssq[threadIdx.x] += ssq[threadIdx.x + st]; }
        __syncthreads();
    }
    if (threadIdx.x == 0) {
        const float N = (float)(HH * WW);
        float mean = ssum[0] / N;
        float var  = fmaxf((ssq[0] - N * mean * mean) / (N - 1.f), 0.f);
        std_ws[b] = sqrtf(var);
    }
}

// ---------------------------------------------------------------- mixture softmax over 1600 px per (b,k)
__global__ __launch_bounds__(256) void pcae_mix(const float* __restrict__ tt,
                                                const float* __restrict__ dm_ws,
                                                float* __restrict__ mix) {
    const int bk = blockIdx.x;
    const float dm = dm_ws[bk];
    const float* t = tt + (size_t)bk * (HH * WW);
    __shared__ float red[256];
    float mx = -1e30f;
    for (int i = threadIdx.x; i < HH * WW; i += 256) mx = fmaxf(mx, dm * t[i]);
    red[threadIdx.x] = mx; __syncthreads();
    for (int st = 128; st > 0; st >>= 1) {
        if (threadIdx.x < st) red[threadIdx.x] = fmaxf(red[threadIdx.x], red[threadIdx.x + st]);
        __syncthreads();
    }
    mx = red[0]; __syncthreads();
    float s = 0.f;
    for (int i = threadIdx.x; i < HH * WW; i += 256) s += __expf(dm * t[i] - mx);
    red[threadIdx.x] = s; __syncthreads();
    for (int st = 128; st > 0; st >>= 1) {
        if (threadIdx.x < st) red[threadIdx.x] += red[threadIdx.x + st];
        __syncthreads();
    }
    float inv = 1.f / red[0];
    for (int i = threadIdx.x; i < HH * WW; i += 256)
        mix[(size_t)bk * (HH * WW) + i] = __expf(dm * t[i] - mx) * inv;
}

// ---------------------------------------------------------------- per-sample log-likelihood partial
__global__ __launch_bounds__(256) void pcae_ll_partial(const float* __restrict__ x,
                                                       const float* __restrict__ tt,
                                                       const float* __restrict__ mix,
                                                       const float* __restrict__ std_ws,
                                                       float* __restrict__ llp) {
    const int b = blockIdx.x;
    const float sd = std_ws[b];
    const float mult = rsqrtf(sd * 2.f * 3.14159265358979f);
    const float pw = -1.f / (2.f * sd * sd);
    float local = 0.f;
    for (int pix = threadIdx.x; pix < HH * WW; pix += 256) {
        float xv = x[(size_t)b * (HH * WW) + pix];
        float s = 0.f;
        for (int k = 0; k < KCAP; ++k) {
            size_t off = ((size_t)b * KCAP + k) * (HH * WW) + pix;
            float df = xv - tt[off];
            s += mult * __expf(df * df * pw) * mix[off];
        }
        local += __logf(s);
    }
    __shared__ float red[256];
    red[threadIdx.x] = local; __syncthreads();
    for (int st = 128; st > 0; st >>= 1) {
        if (threadIdx.x < st) red[threadIdx.x] += red[threadIdx.x + st];
        __syncthreads();
    }
    if (threadIdx.x == 0) llp[b] = red[0];
}

// ---------------------------------------------------------------- deterministic final mean over B
__global__ void pcae_ll_final(const float* __restrict__ llp, float* __restrict__ d_out) {
    if (threadIdx.x == 0) {
        float s = 0.f;
        for (int b = 0; b < BN; ++b) s += llp[b];
        d_out[0] = s / (float)BN;
    }
}

// ================================================================ host
extern "C" void kernel_launch(void* const* d_in, const int* in_sizes, int n_in,
                              void* d_out, int out_size, void* d_ws, size_t ws_size,
                              hipStream_t stream) {
    (void)in_sizes; (void)n_in; (void)out_size; (void)ws_size;
    const float* x   = (const float*)d_in[0];
    const float* w1  = (const float*)d_in[1];
    const float* b1  = (const float*)d_in[2];
    const float* w2  = (const float*)d_in[3];
    const float* b2  = (const float*)d_in[4];
    const float* w3  = (const float*)d_in[5];
    const float* b3  = (const float*)d_in[6];
    const float* w4  = (const float*)d_in[7];
    const float* b4  = (const float*)d_in[8];
    const float* w5  = (const float*)d_in[9];
    const float* b5  = (const float*)d_in[10];
    const float* tpl = (const float*)d_in[11];
    float* out = (float*)d_out;

    char* ws = (char*)d_ws;
    size_t off = 0;
    auto take = [&](size_t bytes) -> char* {
        char* p = ws + off;
        off = (off + bytes + 255) & ~(size_t)255;
        return p;
    };
    const size_t NW34 = (size_t)KCAP * 128 * 128 * 9;   // 3,538,944
    const size_t NW5  = (size_t)KCAP * 16 * 128;        // 49,152
    _Float16* wf2  = (_Float16*)take(NW34 * 2);
    _Float16* wf3  = (_Float16*)take(NW34 * 2);
    _Float16* wf4  = (_Float16*)take(NW34 * 2);
    _Float16* wf5  = (_Float16*)take(NW5 * 2);
    _Float16* act1 = (_Float16*)take((size_t)KCAP * BN * 128 * 361 * 2);
    _Float16* act2 = (_Float16*)take((size_t)KCAP * BN * 128 * 81 * 2);
    _Float16* act3 = (_Float16*)take((size_t)KCAP * BN * 128 * 49 * 2);
    _Float16* act4 = (_Float16*)take((size_t)KCAP * BN * 128 * 25 * 2);   // channel-last [K,B,25,128]
    float*    out5 = (float*)take((size_t)KCAP * BN * 16 * 25 * 4);
    float*    tt   = (float*)take((size_t)BN * KCAP * HH * WW * 4);
    float*    mixb = (float*)take((size_t)BN * KCAP * HH * WW * 4);
    float*    pose = (float*)take((size_t)BN * KCAP * 6 * 4);
    float*    dmw  = (float*)take((size_t)BN * KCAP * 4);
    float*    stdw = (float*)take((size_t)BN * 4);
    float*    llp  = (float*)take((size_t)BN * 4);

    // weight down-conversion
    pcae_cast_f16<<<(int)((NW34 + 255) / 256), 256, 0, stream>>>(w2, wf2, (int)NW34);
    pcae_cast_f16<<<(int)((NW34 + 255) / 256), 256, 0, stream>>>(w3, wf3, (int)NW34);
    pcae_cast_f16<<<(int)((NW34 + 255) / 256), 256, 0, stream>>>(w4, wf4, (int)NW34);
    pcae_cast_f16<<<(int)((NW5 + 255) / 256), 256, 0, stream>>>(w5, wf5, (int)NW5);

    // conv stack
    pcae_conv1<<<dim3(KCAP * BN), 256, 0, stream>>>(x, w1, b1, act1);
    pcae_conv3x3_wmma<2, 19, 19, 9, 9, false><<<dim3(81, KCAP), 256, 0, stream>>>(act1, wf2, b2, act2);
    pcae_conv3x3_wmma<1, 9, 9, 7, 7, false><<<dim3(49, KCAP), 256, 0, stream>>>(act2, wf3, b3, act3);
    pcae_conv3x3_wmma<1, 7, 7, 5, 5, true><<<dim3(25, KCAP), 256, 0, stream>>>(act3, wf4, b4, act4);
    pcae_conv1x1_wmma<<<dim3(13, KCAP), 256, 0, stream>>>(act4, wf5, b5, out5);

    // tail
    pcae_attention<<<(KCAP * BN + 255) / 256, 256, 0, stream>>>(out5, tpl, out, pose, dmw);
    pcae_warp<<<BN * KCAP, 256, 0, stream>>>(pose, tpl, tt);
    pcae_std<<<BN, 256, 0, stream>>>(x, stdw);
    pcae_mix<<<BN * KCAP, 256, 0, stream>>>(tt, dmw, mixb);
    pcae_ll_partial<<<BN, 256, 0, stream>>>(x, tt, mixb, stdw, llp);
    pcae_ll_final<<<1, 64, 0, stream>>>(llp, out);
}
